// DlrmSmall_21869973471264
// MI455X (gfx1250) — compile-verified, matrix operand order checked by reading, not run on hardware
//
#include <hip/hip_runtime.h>
#include <hip/hip_fp16.h>
#include <stdint.h>

typedef __attribute__((ext_vector_type(16))) _Float16 v16h;
typedef __attribute__((ext_vector_type(8)))  _Float16 v8h;
typedef __attribute__((ext_vector_type(8)))  float    v8f;

union ABFrag { v16h v; v8h h[2]; };

#define DLRM_BATCH 16384
#define DLRM_VOCAB 100000
#define DLRM_NSP   26
#define DLRM_NF    27   // 26 sparse + bottom-MLP output

#define USE_ASYNC_LDS 1

#if USE_ASYNC_LDS
// CDNA5 async global->LDS copy, 16 bytes per lane (ASYNCcnt-tracked).
// Generic LDS pointers carry the LDS offset in their low 32 bits (ISA 10.2).
__device__ __forceinline__ void async_cp16(void* lds_dst, const void* gsrc) {
    uint32_t l = (uint32_t)(uintptr_t)lds_dst;
    uint64_t g = (uint64_t)(uintptr_t)gsrc;
    asm volatile("global_load_async_to_lds_b128 %0, %1, off"
                 :: "v"(l), "v"(g) : "memory");
}
__device__ __forceinline__ void wait_async0() {
    asm volatile("s_wait_asynccnt 0" ::: "memory");
}
#endif

// ---------------------------------------------------------------------------
// Weight convert: f32 W[K][N] row-major -> f16 WT[N][Kpad] (transposed, padded)
// ---------------------------------------------------------------------------
__global__ void wconv_kernel(const float* __restrict__ W, _Float16* __restrict__ WT,
                             int K, int N, int Kpad) {
    int idx = blockIdx.x * blockDim.x + threadIdx.x;
    if (idx >= N * Kpad) return;
    int n = idx / Kpad;
    int k = idx - n * Kpad;
    WT[idx] = (k < K) ? (_Float16)W[(size_t)k * N + n] : (_Float16)0.f;
}

// dense f32 [B,13] -> f16 [B,32] zero-padded
__global__ void dconv_kernel(const float* __restrict__ dense, _Float16* __restrict__ h0) {
    int idx = blockIdx.x * blockDim.x + threadIdx.x;
    if (idx >= DLRM_BATCH * 32) return;
    int b = idx >> 5, k = idx & 31;
    h0[idx] = (k < 13) ? (_Float16)dense[b * 13 + k] : (_Float16)0.f;
}

// ---------------------------------------------------------------------------
// Tiled WMMA GEMM: Out[M,N] = relu(A[M,K] * W[K,N] + bias), f16 in/out, f32 acc
// A: f16 [M, lda] row-major. WT: f16 [N, ldw] (transposed weights, ldw = Kpad).
// Block = 256 threads (8 waves). Block tile 128x128, K-step 32.
// Double-buffered LDS tiles staged with global_load_async_to_lds_b128.
// ---------------------------------------------------------------------------
__global__ __launch_bounds__(256) void gemm_wmma_bias_relu(
    const _Float16* __restrict__ A, int lda,
    const _Float16* __restrict__ WT, int ldw,
    const float* __restrict__ bias,
    _Float16* __restrict__ Out, int ldo,
    int K, int relu)
{
#if USE_ASYNC_LDS
    __shared__ __align__(16) _Float16 As[2][128][32];
    __shared__ __align__(16) _Float16 Bs[2][128][32];
#else
    __shared__ __align__(16) _Float16 As[1][128][32];
    __shared__ __align__(16) _Float16 Bs[1][128][32];
#endif

    const int t    = threadIdx.x;
    const int wave = t >> 5;
    const int lane = t & 31;
    const int m0   = blockIdx.x * 128;
    const int n0   = blockIdx.y * 128;

    v8f acc[8] = {};

    const int lrow = t >> 2;         // 0..63 (x2 iterations -> 128 rows)
    const int lcol = (t & 3) * 8;    // 0,8,16,24

    const int lm = lane & 15;
    const int ks = (lane >> 4) * 8;

#if USE_ASYNC_LDS
    auto stage = [&](int buf, int kk) {
#pragma unroll
        for (int i = 0; i < 2; ++i) {
            int row = lrow + i * 64;
            async_cp16(&As[buf][row][lcol], &A[(size_t)(m0 + row) * lda + kk + lcol]);
            async_cp16(&Bs[buf][row][lcol], &WT[(size_t)(n0 + row) * ldw + kk + lcol]);
        }
    };
    stage(0, 0);
    wait_async0();
    __syncthreads();
#endif

    for (int kk = 0; kk < K; kk += 32) {
#if USE_ASYNC_LDS
        const int cur = (kk >> 5) & 1;
        if (kk + 32 < K) stage(cur ^ 1, kk + 32);   // prefetch next tile async
#else
        const int cur = 0;
#pragma unroll
        for (int i = 0; i < 2; ++i) {
            int row = lrow + i * 64;
            *(v8h*)&As[0][row][lcol] =
                *(const v8h*)&A[(size_t)(m0 + row) * lda + kk + lcol];
            *(v8h*)&Bs[0][row][lcol] =
                *(const v8h*)&WT[(size_t)(n0 + row) * ldw + kk + lcol];
        }
        if (kk + 32 < K) {
            __builtin_prefetch(&A[(size_t)(m0 + lrow) * lda + kk + 32 + lcol], 0, 0);
            __builtin_prefetch(&WT[(size_t)(n0 + lrow) * ldw + kk + 32 + lcol], 0, 0);
        }
        __syncthreads();
#endif

        // A fragment: 16-bit 16x32 layout. lane&15 = M row; lane>>4 selects K
        // halves: halfs 0..7 = K[ks..ks+7], halfs 8..15 = K[ks+16..ks+23].
        ABFrag af;
        af.h[0] = *(const v8h*)&As[cur][wave * 16 + lm][ks];
        af.h[1] = *(const v8h*)&As[cur][wave * 16 + lm][ks + 16];
#pragma unroll
        for (int nt = 0; nt < 8; ++nt) {
            // B fragment: B[k][n] = WT[n][k] -> contiguous in WT rows.
            ABFrag bf;
            bf.h[0] = *(const v8h*)&Bs[cur][nt * 16 + lm][ks];
            bf.h[1] = *(const v8h*)&Bs[cur][nt * 16 + lm][ks + 16];
            acc[nt] = __builtin_amdgcn_wmma_f32_16x16x32_f16(
                false, af.v, false, bf.v, (short)0, acc[nt], false, false);
        }

#if USE_ASYNC_LDS
        wait_async0();       // next tile resident before anyone reads it
        __syncthreads();     // and all waves done reading current tile
#else
        __syncthreads();
#endif
    }

    // C/D layout: lanes 0-15 N=lane, VGPR r -> M=r; lanes 16-31 -> M=8+r.
    const int nloc = lane & 15;
    const int mb   = (lane >> 4) * 8;
#pragma unroll
    for (int nt = 0; nt < 8; ++nt) {
        const int col = n0 + nt * 16 + nloc;
        const float bz = bias[col];
#pragma unroll
        for (int r = 0; r < 8; ++r) {
            float v = acc[nt][r] + bz;
            if (relu) v = v > 0.f ? v : 0.f;
            Out[(size_t)(m0 + wave * 16 + mb + r) * ldo + col] = (_Float16)v;
        }
    }
}

// ---------------------------------------------------------------------------
// Interaction: per batch row, gather 26 embedding rows + bot_out into a
// zero-padded 32x128 f16 tile in LDS, compute 32x32 Gram = F*F^T with WMMA,
// emit top_in[row][0:512] = [bot_out(128) | triu(378) | zeros(6)].
// One wave per row, 4 waves per block.
// ---------------------------------------------------------------------------
__global__ __launch_bounds__(128) void interact_wmma(
    const _Float16* __restrict__ botout,   // [B,128] f16
    const int* __restrict__ cat,           // [B,26]
    const float* __restrict__ emb,         // [26*VOCAB,128] f32
    _Float16* __restrict__ topin)          // [B,512] f16
{
    __shared__ __align__(16) _Float16 F[4][32][128];
    __shared__ __align__(16) float    G[4][32][32];

    const int wave = threadIdx.x >> 5;
    const int lane = threadIdx.x & 31;
    const int row  = blockIdx.x * 4 + wave;

    // feats row 0 = bottom MLP output
    {
        const _Float16* br = botout + (size_t)row * 128;
#pragma unroll
        for (int j = 0; j < 4; ++j) F[wave][0][lane * 4 + j] = br[lane * 4 + j];
    }
    // feats rows 1..26 = embedding gather (f32 -> f16), float4 per lane
    for (int f = 0; f < DLRM_NSP; ++f) {
        const int c = cat[(size_t)row * DLRM_NSP + f];
        const float4 v = *(const float4*)(emb + ((size_t)f * DLRM_VOCAB + (size_t)c) * 128
                                          + lane * 4);
        F[wave][1 + f][lane * 4 + 0] = (_Float16)v.x;
        F[wave][1 + f][lane * 4 + 1] = (_Float16)v.y;
        F[wave][1 + f][lane * 4 + 2] = (_Float16)v.z;
        F[wave][1 + f][lane * 4 + 3] = (_Float16)v.w;
    }
    // zero-pad rows 27..31
#pragma unroll
    for (int rr = DLRM_NF; rr < 32; ++rr)
#pragma unroll
        for (int j = 0; j < 4; ++j) F[wave][rr][lane * 4 + j] = (_Float16)0.f;
    __syncthreads();

    // Gram = F (32x128) * F^T : 2x2 tiles of 16x16, K loop of 4.
    // B-matrix (K x N) fragment of F^T equals the A fragment of tile n.
    v8f a00 = {}, a01 = {}, a10 = {}, a11 = {};
    const int lm = lane & 15;
    const int ks = (lane >> 4) * 8;
#pragma unroll
    for (int kk = 0; kk < 128; kk += 32) {
        ABFrag f0, f1;
        f0.h[0] = *(const v8h*)&F[wave][lm][kk + ks];
        f0.h[1] = *(const v8h*)&F[wave][lm][kk + ks + 16];
        f1.h[0] = *(const v8h*)&F[wave][16 + lm][kk + ks];
        f1.h[1] = *(const v8h*)&F[wave][16 + lm][kk + ks + 16];
        a00 = __builtin_amdgcn_wmma_f32_16x16x32_f16(false, f0.v, false, f0.v, (short)0, a00, false, false);
        a01 = __builtin_amdgcn_wmma_f32_16x16x32_f16(false, f0.v, false, f1.v, (short)0, a01, false, false);
        a10 = __builtin_amdgcn_wmma_f32_16x16x32_f16(false, f1.v, false, f0.v, (short)0, a10, false, false);
        a11 = __builtin_amdgcn_wmma_f32_16x16x32_f16(false, f1.v, false, f1.v, (short)0, a11, false, false);
    }

    const int nloc = lane & 15;
    const int mb   = (lane >> 4) * 8;
#pragma unroll
    for (int r = 0; r < 8; ++r) {
        G[wave][mb + r][nloc]           = a00[r];
        G[wave][mb + r][16 + nloc]      = a01[r];
        G[wave][16 + mb + r][nloc]      = a10[r];
        G[wave][16 + mb + r][16 + nloc] = a11[r];
    }
    __syncthreads();

    _Float16* o = topin + (size_t)row * 512;
    for (int j = lane; j < 128; j += 32) o[j] = F[wave][0][j];        // bot_out
    for (int idx = lane; idx < 378; idx += 32) {                      // triu(27)
        int i = 0, rem = idx;
        while (rem >= DLRM_NF - i) { rem -= DLRM_NF - i; ++i; }
        o[128 + idx] = (_Float16)G[wave][i][i + rem];
    }
    if (lane < 6) o[506 + lane] = (_Float16)0.f;                      // pad
}

// Final 256 -> 1 layer, f32
__global__ void final_dot(const _Float16* __restrict__ a, const float* __restrict__ w,
                          const float* __restrict__ b, float* __restrict__ out) {
    int r = blockIdx.x * blockDim.x + threadIdx.x;
    if (r >= DLRM_BATCH) return;
    const _Float16* ar = a + (size_t)r * 256;
    float s = b[0];
#pragma unroll 8
    for (int k = 0; k < 256; ++k) s += (float)ar[k] * w[k];
    out[r] = s;
}

// ---------------------------------------------------------------------------
extern "C" void kernel_launch(void* const* d_in, const int* in_sizes, int n_in,
                              void* d_out, int out_size, void* d_ws, size_t ws_size,
                              hipStream_t stream) {
    const float* dense = (const float*)d_in[0];
    const int*   cat   = (const int*)d_in[1];
    const float* emb   = (const float*)d_in[2];
    const float* bw[3] = {(const float*)d_in[3], (const float*)d_in[5], (const float*)d_in[7]};
    const float* bb[3] = {(const float*)d_in[4], (const float*)d_in[6], (const float*)d_in[8]};
    const float* tw[5] = {(const float*)d_in[9],  (const float*)d_in[11], (const float*)d_in[13],
                          (const float*)d_in[15], (const float*)d_in[17]};
    const float* tb[5] = {(const float*)d_in[10], (const float*)d_in[12], (const float*)d_in[14],
                          (const float*)d_in[16], (const float*)d_in[18]};
    float* out = (float*)d_out;

    char* ws = (char*)d_ws;
    size_t off = 0;
    auto take = [&](size_t elems) {
        void* p = ws + off;
        off += (elems * sizeof(_Float16) + 255) & ~(size_t)255;
        return (_Float16*)p;
    };
    _Float16* h0   = take((size_t)DLRM_BATCH * 32);
    _Float16* a1   = take((size_t)DLRM_BATCH * 512);
    _Float16* a2   = take((size_t)DLRM_BATCH * 256);
    _Float16* bo   = take((size_t)DLRM_BATCH * 128);
    _Float16* tin  = take((size_t)DLRM_BATCH * 512);
    _Float16* t1   = take((size_t)DLRM_BATCH * 1024);
    _Float16* t2   = take((size_t)DLRM_BATCH * 1024);
    _Float16* t3   = take((size_t)DLRM_BATCH * 512);
    _Float16* t4   = take((size_t)DLRM_BATCH * 256);
    _Float16* bw0t = take(512 * 32);
    _Float16* bw1t = take(256 * 512);
    _Float16* bw2t = take(128 * 256);
    _Float16* tw0t = take(1024 * 512);
    _Float16* tw1t = take(1024 * 1024);
    _Float16* tw2t = take(512 * 1024);
    _Float16* tw3t = take(256 * 512);

    auto wc = [&](const float* W, _Float16* WT, int K, int N, int Kpad) {
        int total = N * Kpad;
        wconv_kernel<<<(total + 255) / 256, 256, 0, stream>>>(W, WT, K, N, Kpad);
    };
    wc(bw[0], bw0t, 13,   512,  32);
    wc(bw[1], bw1t, 512,  256,  512);
    wc(bw[2], bw2t, 256,  128,  256);
    wc(tw[0], tw0t, 506,  1024, 512);
    wc(tw[1], tw1t, 1024, 1024, 1024);
    wc(tw[2], tw2t, 1024, 512,  1024);
    wc(tw[3], tw3t, 512,  256,  512);

    dconv_kernel<<<(DLRM_BATCH * 32 + 255) / 256, 256, 0, stream>>>(dense, h0);

    auto gm = [&](const _Float16* A, int lda, const _Float16* WT, int ldw,
                  const float* bias, _Float16* O, int N, int K, int relu) {
        dim3 grid(DLRM_BATCH / 128, N / 128);
        gemm_wmma_bias_relu<<<grid, dim3(256), 0, stream>>>(A, lda, WT, ldw, bias, O, N, K, relu);
    };

    // Bottom MLP
    gm(h0, 32,  bw0t, 32,  bb[0], a1, 512, 32,  1);
    gm(a1, 512, bw1t, 512, bb[1], a2, 256, 512, 1);
    gm(a2, 256, bw2t, 256, bb[2], bo, 128, 256, 1);
    // Embedding gather + pairwise interaction
    interact_wmma<<<DLRM_BATCH / 4, 128, 0, stream>>>(bo, cat, emb, tin);
    // Top MLP
    gm(tin, 512,  tw0t, 512,  tb[0], t1, 1024, 512,  1);
    gm(t1,  1024, tw1t, 1024, tb[1], t2, 1024, 1024, 1);
    gm(t2,  1024, tw2t, 1024, tb[2], t3, 512,  1024, 1);
    gm(t3,  512,  tw3t, 512,  tb[3], t4, 256,  512,  1);
    final_dot<<<DLRM_BATCH / 256, 256, 0, stream>>>(t4, tw[4], tb[4], out);

    (void)in_sizes; (void)n_in; (void)out_size; (void)ws_size;
}
